// SelfAttention_704374636936
// MI455X (gfx1250) — matrix operand order repaired and checked by reading, not compiled, not run on hardware
//
#include <hip/hip_runtime.h>
#include <hip/hip_bf16.h>
#include <stdint.h>

typedef float  v8f   __attribute__((ext_vector_type(8)));
typedef float  v2f   __attribute__((ext_vector_type(2)));
typedef __bf16 v16bf __attribute__((ext_vector_type(16)));

#define BATCH 8
#define CCH   64
#define C8    8
#define NPIX  4096
#define KT    64      // keys per iteration
#define NQ    128     // query rows per workgroup (8 waves x 16)
#define SKSTR 72      // K-tile LDS row stride (floats)
#define VCP   68      // packed-V LDS row stride (dwords)
#define PSTR  72      // P LDS row stride (bf16 elems), 144B rows -> 16B aligned
#define OSTR  65      // output staging stride (floats)

// smem layout (bytes): sK0 @0 (2304), sK1 @2304, sV0 @4608 (8704),
//                      sV1 @13312,    sP @22016 (18432)  -> total 40448
// sO (128*65*4 = 33280) overlays the front after the loop.
#define OFF_K0 0
#define KSIZE  2304
#define OFF_V0 4608
#define VSIZE  8704
#define OFF_P  22016
#define SMEM_BYTES 40448

__device__ __forceinline__ unsigned int bf16_bits(float f) {
    unsigned int u = __float_as_uint(f);
    u += 0x7FFFu + ((u >> 16) & 1u);   // round-to-nearest-even
    return u >> 16;
}
__device__ __forceinline__ unsigned int pack_bf16(float lo, float hi) {
    return bf16_bits(lo) | (bf16_bits(hi) << 16);
}

// xor-butterfly max within 16-lane rows using DPP ROW_XMASK (pure VALU)
template <int MASK>
__device__ __forceinline__ float xmax16(float v) {
    int t = __builtin_amdgcn_mov_dpp(__float_as_int(v), 0x160 | MASK, 0xf, 0xf, true);
    return fmaxf(v, __int_as_float(t));
}

// ---------------------------------------------------------------------------
// Phase 1: per-pixel 1x1-conv projections.  q:(B,N,8)  k:(B,8,N)  vT:(B,N,64)
// ---------------------------------------------------------------------------
__global__ __launch_bounds__(256) void proj_qkv(
    const float* __restrict__ x,
    const float* __restrict__ wq, const float* __restrict__ bq,
    const float* __restrict__ wk, const float* __restrict__ bk,
    const float* __restrict__ wv, const float* __restrict__ bv,
    float* __restrict__ q, float* __restrict__ k, float* __restrict__ vT)
{
    __shared__ float sW[80 * 64];   // rows 0-7 wq, 8-15 wk, 16-79 wv  (o-major)
    __shared__ float sB[80];
    const int tid = threadIdx.x;
    for (int i = tid; i < 8 * 64; i += 256)  sW[i] = wq[i];
    for (int i = tid; i < 8 * 64; i += 256)  sW[8 * 64 + i] = wk[i];
    for (int i = tid; i < 64 * 64; i += 256) sW[16 * 64 + i] = wv[i];
    if (tid < 8)        sB[tid] = bq[tid];
    else if (tid < 16)  sB[tid] = bk[tid - 8];
    else if (tid < 80)  sB[tid] = bv[tid - 16];
    __syncthreads();

    const int b = blockIdx.y;
    const int n = blockIdx.x * 256 + tid;

    float acc[80];
#pragma unroll
    for (int o = 0; o < 80; ++o) acc[o] = sB[o];

    const float* xp = x + (size_t)b * CCH * NPIX + n;
    for (int c = 0; c < CCH; ++c) {
        const float xc = xp[(size_t)c * NPIX];
        const float* wc = sW + c;                       // uniform (broadcast) reads
#pragma unroll
        for (int o = 0; o < 80; ++o) acc[o] = fmaf(wc[o * 64], xc, acc[o]);
    }

    float* qp = q + ((size_t)b * NPIX + n) * C8;
#pragma unroll
    for (int t = 0; t < 2; ++t)
        ((float4*)qp)[t] = make_float4(acc[4*t], acc[4*t+1], acc[4*t+2], acc[4*t+3]);
#pragma unroll
    for (int o = 0; o < C8; ++o) k[((size_t)b * C8 + o) * NPIX + n] = acc[8 + o];
    float* vp = vT + ((size_t)b * NPIX + n) * CCH;
#pragma unroll
    for (int t = 0; t < 16; ++t)
        ((float4*)vp)[t] = make_float4(acc[16+4*t], acc[16+4*t+1], acc[16+4*t+2], acc[16+4*t+3]);
}

// ---------------------------------------------------------------------------
// Phase 2: fused flash attention, double-buffered K/V tiles.
//   S  : v_wmma_f32_16x16x4_f32   (fp32 logits, K-dim = 8)
//   P*V: v_wmma_f32_16x16x32_bf16 (bf16 inputs, fp32 accumulate)
//   l  : accumulated as a "ones column" WMMA (no shuffle reduction)
// ---------------------------------------------------------------------------
__global__ __launch_bounds__(256) void flash_attn(
    const float* __restrict__ x,
    const float* __restrict__ q,
    const float* __restrict__ kmat,
    const float* __restrict__ vT,
    const float* __restrict__ gamma,
    float* __restrict__ out)
{
    __shared__ __align__(16) char smem[SMEM_BYTES];
    unsigned short* sP = (unsigned short*)(smem + OFF_P);
    float*          sO = (float*)smem;                  // reused after loop

    const int tid  = threadIdx.x;
    const int wid  = tid >> 5;
    const int lane = tid & 31;
    const int h    = lane >> 4;
    const int ln   = lane & 15;
    const int b     = blockIdx.y;
    const int qbase = blockIdx.x * NQ;

    // coop-load index precompute
    const int ko0 = tid >> 6,         kj0 = tid & 63;          // K elems tid, tid+256
    const int ko1 = (tid + 256) >> 6, kj1 = tid & 63;
    const int jp0 = tid >> 4,         c40 = tid & 15;          // V pair-chunks
    const int jp1 = (tid + 256) >> 4, c41 = tid & 15;
    const float* kbase0 = kmat + ((size_t)b * C8 + ko0) * NPIX + kj0;
    const float* kbase1 = kmat + ((size_t)b * C8 + ko1) * NPIX + kj1;
    const float* vbase  = vT + (size_t)b * NPIX * CCH;

    // --- Q A-fragments (f32 16x4 layout), loaded once ---
    const float* qrow = q + ((size_t)b * NPIX + qbase + wid * 16 + ln) * C8;
    v2f a0, a1;
    a0.x = qrow[2 * h];     a0.y = qrow[2 * h + 1];
    a1.x = qrow[4 + 2 * h]; a1.y = qrow[4 + 2 * h + 1];

    // --- ones-column B fragment (col 0 = 1.0bf16 pairs) for row sums ---
    union { unsigned int u[8]; v16bf v; } onesB;
    {
        unsigned int ov = (ln == 0) ? 0x3F803F80u : 0u;
#pragma unroll
        for (int t = 0; t < 8; ++t) onesB.u[t] = ov;
    }

    v8f Oc[4], Ol;
    float m[8];
#pragma unroll
    for (int cb = 0; cb < 4; ++cb)
#pragma unroll
        for (int r = 0; r < 8; ++r) Oc[cb][r] = 0.0f;
#pragma unroll
    for (int r = 0; r < 8; ++r) { Ol[r] = 0.0f; m[r] = -3.0e38f; }

    // staging registers for the next tile
    float  kr0, kr1;
    float4 va0, vb0, va1, vb1;

    // ---- prologue: fetch tile 0 and commit to buffer 0 ----
    kr0 = kbase0[0]; kr1 = kbase1[0];
    va0 = *(const float4*)(vbase + (size_t)(2 * jp0)     * CCH + 4 * c40);
    vb0 = *(const float4*)(vbase + (size_t)(2 * jp0 + 1) * CCH + 4 * c40);
    va1 = *(const float4*)(vbase + (size_t)(2 * jp1)     * CCH + 4 * c41);
    vb1 = *(const float4*)(vbase + (size_t)(2 * jp1 + 1) * CCH + 4 * c41);
    {
        float*        nK = (float*)(smem + OFF_K0);
        unsigned int* nV = (unsigned int*)(smem + OFF_V0);
        nK[ko0 * SKSTR + kj0] = kr0;
        nK[ko1 * SKSTR + kj1] = kr1;
        uint4 p0 = make_uint4(pack_bf16(va0.x, vb0.x), pack_bf16(va0.y, vb0.y),
                              pack_bf16(va0.z, vb0.z), pack_bf16(va0.w, vb0.w));
        uint4 p1 = make_uint4(pack_bf16(va1.x, vb1.x), pack_bf16(va1.y, vb1.y),
                              pack_bf16(va1.z, vb1.z), pack_bf16(va1.w, vb1.w));
        *(uint4*)(nV + jp0 * VCP + 4 * c40) = p0;
        *(uint4*)(nV + jp1 * VCP + 4 * c41) = p1;
    }
    __syncthreads();

    for (int kb = 0; kb < NPIX; kb += KT) {
        const int  cur  = (kb >> 6) & 1;
        const bool more = (kb + KT) < NPIX;

        // ---- issue next tile's global loads (latency hidden by compute) ----
        if (more) {
            const int nkb = kb + KT;
            kr0 = kbase0[nkb]; kr1 = kbase1[nkb];
            va0 = *(const float4*)(vbase + (size_t)(nkb + 2 * jp0)     * CCH + 4 * c40);
            vb0 = *(const float4*)(vbase + (size_t)(nkb + 2 * jp0 + 1) * CCH + 4 * c40);
            va1 = *(const float4*)(vbase + (size_t)(nkb + 2 * jp1)     * CCH + 4 * c41);
            vb1 = *(const float4*)(vbase + (size_t)(nkb + 2 * jp1 + 1) * CCH + 4 * c41);
        }

        // ---- S = Q K^T : 16 x KT fp32 ----
        const float* sKc = (const float*)(smem + OFF_K0 + cur * KSIZE);
        v8f s[4];
#pragma unroll
        for (int jj = 0; jj < 4; ++jj) {
            const float* kc = sKc + jj * 16 + ln;
            v2f b0, b1;
            b0.x = kc[(2 * h) * SKSTR];     b0.y = kc[(2 * h + 1) * SKSTR];
            b1.x = kc[(4 + 2 * h) * SKSTR]; b1.y = kc[(4 + 2 * h + 1) * SKSTR];
            v8f z = {};
            z = __builtin_amdgcn_wmma_f32_16x16x4_f32(false, a0, false, b0,
                                                      (short)0, z, false, false);
            s[jj] = __builtin_amdgcn_wmma_f32_16x16x4_f32(false, a1, false, b1,
                                                          (short)0, z, false, false);
        }

        // ---- online max: in-reg max over 4 blocks, then DPP xor butterfly ----
        float mn[8];
#pragma unroll
        for (int r = 0; r < 8; ++r) {
            float v = fmaxf(fmaxf(s[0][r], s[1][r]), fmaxf(s[2][r], s[3][r]));
            mn[r] = fmaxf(v, m[r]);
        }
#pragma unroll
        for (int r = 0; r < 8; ++r) mn[r] = xmax16<1>(mn[r]);
#pragma unroll
        for (int r = 0; r < 8; ++r) mn[r] = xmax16<2>(mn[r]);
#pragma unroll
        for (int r = 0; r < 8; ++r) mn[r] = xmax16<4>(mn[r]);
#pragma unroll
        for (int r = 0; r < 8; ++r) mn[r] = xmax16<8>(mn[r]);

#pragma unroll
        for (int r = 0; r < 8; ++r) {
            float sc = __expf(m[r] - mn[r]);
            m[r] = mn[r];
            Ol[r] *= sc;
#pragma unroll
            for (int cb = 0; cb < 4; ++cb) Oc[cb][r] *= sc;
        }

        // ---- P = exp(S-m) -> bf16 rows in LDS (C/D layout -> row-major) ----
#pragma unroll
        for (int jj = 0; jj < 4; ++jj)
#pragma unroll
            for (int r = 0; r < 8; ++r) {
                float p = __expf(s[jj][r] - m[r]);
                sP[(wid * 16 + r + 8 * h) * PSTR + jj * 16 + ln] =
                    (unsigned short)bf16_bits(p);
            }
        asm volatile("" ::: "memory");   // keep u16 stores before u32 frag loads

        // ---- A-fragments of P (bf16 16x32 layout): 16B LDS loads ----
        v16bf A[2];
#pragma unroll
        for (int kf = 0; kf < 2; ++kf) {
            const unsigned int* pr = (const unsigned int*)(sP + (wid * 16 + ln) * PSTR);
            int d0 = kf * 16 + h * 8;
            union { unsigned int u[8]; v16bf v; } uu;
#pragma unroll
            for (int t = 0; t < 4; ++t) uu.u[t] = pr[d0 + t];
#pragma unroll
            for (int t = 0; t < 4; ++t) uu.u[4 + t] = pr[d0 + 4 + t];
            A[kf] = uu.v;
        }

        // ---- O += P V ; l += P 1 ----
        const unsigned int* sVc = (const unsigned int*)(smem + OFF_V0 + cur * VSIZE);
#pragma unroll
        for (int cb = 0; cb < 4; ++cb)
#pragma unroll
            for (int kf = 0; kf < 2; ++kf) {
                union { unsigned int u[8]; v16bf v; } bb;
                const unsigned int* vcol = sVc + (kf * 16 + h * 8) * VCP + cb * 16 + ln;
#pragma unroll
                for (int t = 0; t < 8; ++t) bb.u[t] = vcol[t * VCP];
                Oc[cb] = __builtin_amdgcn_wmma_f32_16x16x32_bf16(
                    false, A[kf], false, bb.v, (short)0, Oc[cb], false, false);
            }
        Ol = __builtin_amdgcn_wmma_f32_16x16x32_bf16(
            false, A[0], false, onesB.v, (short)0, Ol, false, false);
        Ol = __builtin_amdgcn_wmma_f32_16x16x32_bf16(
            false, A[1], false, onesB.v, (short)0, Ol, false, false);

        // ---- commit next tile to the other buffer ----
        if (more) {
            float*        nK = (float*)(smem + OFF_K0 + (cur ^ 1) * KSIZE);
            unsigned int* nV = (unsigned int*)(smem + OFF_V0 + (cur ^ 1) * VSIZE);
            nK[ko0 * SKSTR + kj0] = kr0;
            nK[ko1 * SKSTR + kj1] = kr1;
            uint4 p0 = make_uint4(pack_bf16(va0.x, vb0.x), pack_bf16(va0.y, vb0.y),
                                  pack_bf16(va0.z, vb0.z), pack_bf16(va0.w, vb0.w));
            uint4 p1 = make_uint4(pack_bf16(va1.x, vb1.x), pack_bf16(va1.y, vb1.y),
                                  pack_bf16(va1.z, vb1.z), pack_bf16(va1.w, vb1.w));
            *(uint4*)(nV + jp0 * VCP + 4 * c40) = p0;
            *(uint4*)(nV + jp1 * VCP + 4 * c41) = p1;
        }
        __syncthreads();
    }

    // ---- broadcast row sums (col 0 of Ol lives in lane 16h), normalize ----
#pragma unroll
    for (int r = 0; r < 8; ++r) {
        float lsum = __shfl(Ol[r], 16 * h, 32);
        float invl = 1.0f / lsum;
        float* orow = sO + (wid * 16 + r + 8 * h) * OSTR + ln;
#pragma unroll
        for (int cb = 0; cb < 4; ++cb) orow[cb * 16] = Oc[cb][r] * invl;
    }
    __syncthreads();

    const float g = gamma[0];
    for (int i = tid; i < NQ * CCH; i += 256) {
        int c = i >> 7, nn = i & (NQ - 1);
        size_t xi = ((size_t)b * CCH + c) * NPIX + qbase + nn;
        out[xi] = fmaf(g, sO[nn * OSTR + c], x[xi]);
    }
}

// ---------------------------------------------------------------------------
extern "C" void kernel_launch(void* const* d_in, const int* in_sizes, int n_in,
                              void* d_out, int out_size, void* d_ws, size_t ws_size,
                              hipStream_t stream) {
    (void)in_sizes; (void)n_in; (void)out_size; (void)ws_size;
    const float* x     = (const float*)d_in[0];
    const float* wq    = (const float*)d_in[1];
    const float* bq    = (const float*)d_in[2];
    const float* wk    = (const float*)d_in[3];
    const float* bk    = (const float*)d_in[4];
    const float* wv    = (const float*)d_in[5];
    const float* bv    = (const float*)d_in[6];
    const float* gamma = (const float*)d_in[7];
    float* out = (float*)d_out;

    float* qW = (float*)d_ws;                       // B*N*8
    float* kW = qW + (size_t)BATCH * NPIX * C8;     // B*8*N
    float* vW = kW + (size_t)BATCH * C8 * NPIX;     // B*N*64

    proj_qkv<<<dim3(NPIX / 256, BATCH), 256, 0, stream>>>(
        x, wq, bq, wk, bk, wv, bv, qW, kW, vW);
    flash_attn<<<dim3(NPIX / NQ, BATCH), 256, 0, stream>>>(
        x, qW, kW, vW, gamma, out);
}
